// MultiHeadAttention_23158463660606
// MI455X (gfx1250) — compile-verified, matrix operand order checked by reading, not run on hardware
//
#include <hip/hip_runtime.h>

// ---------- types ----------
typedef __bf16 bf16;
typedef __attribute__((ext_vector_type(16))) __bf16 v16bf;
typedef __attribute__((ext_vector_type(8)))  __bf16 v8bf;
typedef __attribute__((ext_vector_type(8)))  float  v8f;
typedef __attribute__((ext_vector_type(4)))  float  v4f;
typedef int v4i_ __attribute__((vector_size(16)));   // matches builtin param type

#define T_SEQ   2048
#define D_MODEL 1024
#define H_NUM   16
#define D_K     64

// ---------- CDNA5 async global->LDS path (guarded; fallback = load+ds_store) ----
#if __has_builtin(__builtin_amdgcn_global_load_async_to_lds_b128)
#define HAVE_ASYNC_LDS 1
#else
#define HAVE_ASYNC_LDS 0
#endif

__device__ __forceinline__ void cp16_g2l(const bf16* g, bf16* l) {
#if HAVE_ASYNC_LDS
  __builtin_amdgcn_global_load_async_to_lds_b128(
      (__attribute__((address_space(1))) v4i_*)(unsigned long long)g,
      (__attribute__((address_space(3))) v4i_*)(unsigned int)(unsigned long long)l,
      0, 0);
#else
  *(v8bf*)l = *(const v8bf*)g;
#endif
}

__device__ __forceinline__ void wait_async_lds() {
#if HAVE_ASYNC_LDS
#if __has_builtin(__builtin_amdgcn_s_wait_asynccnt)
  __builtin_amdgcn_s_wait_asynccnt(0);
#else
  asm volatile("s_wait_asynccnt 0" ::: "memory");
#endif
#endif
}

// Load 16 bf16 (two contiguous 16B chunks) as a WMMA fragment.
__device__ __forceinline__ v16bf load_frag_bf16(const bf16* p0, const bf16* p1) {
  v8bf lo = *(const v8bf*)(p0);
  v8bf hi = *(const v8bf*)(p1);
  return __builtin_shufflevector(lo, hi, 0, 1, 2, 3, 4, 5, 6, 7,
                                 8, 9, 10, 11, 12, 13, 14, 15);
}

__device__ __forceinline__ v8f wmma_bf16(v16bf a, v16bf b, v8f c) {
  return __builtin_amdgcn_wmma_f32_16x16x32_bf16(false, a, false, b, (short)0, c,
                                                 false, false);
}

// ---------------------------------------------------------------------------
// Kernel 0: one-time fp32 -> bf16 conversion (vectorized, 8 elems/thread)
// ---------------------------------------------------------------------------
__global__ __launch_bounds__(256) void cvt_bf16_kernel(const float* __restrict__ in,
                                                       bf16* __restrict__ out,
                                                       int n8) {
  const int i = blockIdx.x * blockDim.x + threadIdx.x;
  if (i < n8) {
    const float* p = in + (size_t)i * 8;
    v4f a = *(const v4f*)p;
    v4f b = *(const v4f*)(p + 4);
    v8bf o;
#pragma unroll
    for (int j = 0; j < 4; ++j) { o[j] = (bf16)a[j]; o[4 + j] = (bf16)b[j]; }
    *(v8bf*)(out + (size_t)i * 8) = o;
  }
}

// ---------------------------------------------------------------------------
// Kernel 1: projection  Y = (X @ W^T + b) * scale, all-bf16 operands.
//   mode 0: write [b,h,t,d] (Q with scale=1/8, K)    mode 1: write [b,h,d,t] (V)
// Wave tile 64x32 (4x2 WMMA). Block = 8 waves = 128x128 tile.
// ---------------------------------------------------------------------------
__global__ __launch_bounds__(256) void proj_kernel(
    const bf16* __restrict__ X, const bf16* __restrict__ W,
    const float* __restrict__ bias, bf16* __restrict__ out,
    int mode, float scale) {
  const int lane   = threadIdx.x & 31;
  const int wave   = threadIdx.x >> 5;
  const int n_base = blockIdx.x * 128 + (wave & 3) * 32;
  const int m_base = blockIdx.y * 128 + (wave >> 2) * 64;
  const int ln = lane & 15, kh = lane >> 4;

  v8f acc[4][2] = {};
  for (int k0 = 0; k0 < D_MODEL; k0 += 32) {
    v16bf a[4], b[2];
#pragma unroll
    for (int i = 0; i < 4; ++i) {
      const bf16* pa = X + (size_t)(m_base + i * 16 + ln) * D_MODEL + k0 + kh * 8;
      __builtin_prefetch(pa + 32, 0, 1);
      a[i] = load_frag_bf16(pa, pa + 16);
    }
#pragma unroll
    for (int j = 0; j < 2; ++j) {
      const bf16* pb = W + (size_t)(n_base + j * 16 + ln) * D_MODEL + k0 + kh * 8;
      __builtin_prefetch(pb + 32, 0, 1);
      b[j] = load_frag_bf16(pb, pb + 16);
    }
#pragma unroll
    for (int i = 0; i < 4; ++i)
#pragma unroll
      for (int j = 0; j < 2; ++j) acc[i][j] = wmma_bf16(a[i], b[j], acc[i][j]);
  }

#pragma unroll
  for (int i = 0; i < 4; ++i) {
#pragma unroll
    for (int j = 0; j < 2; ++j) {
      const int n_g = n_base + j * 16 + ln;
      const float bv = bias[n_g];
      const int h = n_g >> 6, d = n_g & 63;
      if (mode == 0) {
#pragma unroll
        for (int r = 0; r < 8; ++r) {
          const int m = m_base + i * 16 + r + 8 * kh;
          const int bb = m >> 11, t = m & (T_SEQ - 1);
          out[((size_t)((bb * H_NUM + h) * T_SEQ + t) << 6) + d] =
              (bf16)((acc[i][j][r] + bv) * scale);
        }
      } else {
        v8bf pk;
#pragma unroll
        for (int r = 0; r < 8; ++r) pk[r] = (bf16)((acc[i][j][r] + bv) * scale);
        const int m0 = m_base + i * 16 + 8 * kh;
        const int bb = m0 >> 11, t = m0 & (T_SEQ - 1);
        *(v8bf*)(out + (size_t)((bb * H_NUM + h) * D_K + d) * T_SEQ + t) = pk;
      }
    }
  }
}

// ---------------------------------------------------------------------------
// Kernel 2: flash attention per (b,h). Q,K in [b,h,t,d]; V in [b,h,d,t].
// Grid (B*H, T/128); 8 waves/block, one 16-row query block per wave.
// K/V chunks (32 keys) staged cooperatively into LDS, double-buffered via the
// async global->LDS engine (ASYNCcnt) when available.
// ---------------------------------------------------------------------------
#define KST 72  // bf16 elems per K-chunk LDS row (144B: 16B-aligned, conflict-free)
#define VST 40  // bf16 elems per V-chunk LDS row (80B)

__global__ __launch_bounds__(256) void flash_attn_kernel(
    const bf16* __restrict__ Q, const bf16* __restrict__ K,
    const bf16* __restrict__ Vt, bf16* __restrict__ ctx) {
  __shared__ __align__(16) bf16 kbuf[2][32 * KST];
  __shared__ __align__(16) bf16 vbuf[2][64 * VST];
  __shared__ __align__(16) bf16 plds[8][16 * 40];

  const int tid  = threadIdx.x;
  const int lane = tid & 31, wave = tid >> 5;
  const int bh = blockIdx.x;
  const int b = bh >> 4, h = bh & 15;
  const int ln = lane & 15, kh = lane >> 4;

  const bf16* Qh = Q  + (size_t)bh * T_SEQ * D_K;
  const bf16* Kh = K  + (size_t)bh * T_SEQ * D_K;
  const bf16* Vh = Vt + (size_t)bh * D_K * T_SEQ;
  bf16* lb = plds[wave];

  const int t0 = blockIdx.y * 128 + wave * 16;

  // cooperative staging coordinates (256 threads x 16B)
  const int krow = tid >> 3, kc = tid & 7;  // K chunk: 32 rows x 128B
  const int vrow = tid >> 2, vc = tid & 3;  // V chunk: 64 rows x 64B

  // Q fragments (scale pre-folded in projection)
  const bf16* qp = Qh + (size_t)(t0 + ln) * D_K + kh * 8;
  const v16bf qa0 = load_frag_bf16(qp, qp + 16);
  const v16bf qa1 = load_frag_bf16(qp + 32, qp + 48);

  v8f o[4] = {};
  float rowmax[8], rowsum[8];
#pragma unroll
  for (int r = 0; r < 8; ++r) { rowmax[r] = -3.0e38f; rowsum[r] = 0.0f; }

  // prologue: stage chunk 0
  cp16_g2l(Kh + (size_t)krow * D_K + kc * 8, &kbuf[0][krow * KST + kc * 8]);
  cp16_g2l(Vh + (size_t)vrow * T_SEQ + vc * 8, &vbuf[0][vrow * VST + vc * 8]);
  wait_async_lds();
  __syncthreads();

  int cur = 0;
  for (int s0 = 0; s0 < T_SEQ; s0 += 32, cur ^= 1) {
    const int nxt = cur ^ 1;
    if (s0 + 32 < T_SEQ) {
      cp16_g2l(Kh + (size_t)(s0 + 32 + krow) * D_K + kc * 8,
               &kbuf[nxt][krow * KST + kc * 8]);
      cp16_g2l(Vh + (size_t)vrow * T_SEQ + (s0 + 32) + vc * 8,
               &vbuf[nxt][vrow * VST + vc * 8]);
    }

    // ---- scores S[16x32] = Q[16x64] @ K^T (from LDS) ----
    v8f sacc[2] = {};
#pragma unroll
    for (int c = 0; c < 2; ++c) {
      const bf16* kp = &kbuf[cur][(c * 16 + ln) * KST + kh * 8];
      const v16bf kb0 = load_frag_bf16(kp, kp + 16);
      const v16bf kb1 = load_frag_bf16(kp + 32, kp + 48);
      sacc[c] = wmma_bf16(qa0, kb0, sacc[c]);
      sacc[c] = wmma_bf16(qa1, kb1, sacc[c]);
    }

    // ---- online softmax (rows r + 8*kh per half-wave) ----
    float mnew[8], alpha[8], psum[8];
#pragma unroll
    for (int r = 0; r < 8; ++r)
      mnew[r] = fmaxf(rowmax[r], fmaxf(sacc[0][r], sacc[1][r]));
#pragma unroll
    for (int mask = 1; mask < 16; mask <<= 1)
#pragma unroll
      for (int r = 0; r < 8; ++r)
        mnew[r] = fmaxf(mnew[r], __shfl_xor(mnew[r], mask, 32));
#pragma unroll
    for (int r = 0; r < 8; ++r) {
      alpha[r] = __expf(rowmax[r] - mnew[r]);
      rowmax[r] = mnew[r];
      psum[r] = 0.0f;
    }
#pragma unroll
    for (int c = 0; c < 2; ++c)
#pragma unroll
      for (int r = 0; r < 8; ++r) {
        const float p = __expf(sacc[c][r] - mnew[r]);
        psum[r] += p;
        lb[(r + 8 * kh) * 40 + c * 16 + ln] = (bf16)p;  // C->A layout round trip
      }
#pragma unroll
    for (int mask = 1; mask < 16; mask <<= 1)
#pragma unroll
      for (int r = 0; r < 8; ++r) psum[r] += __shfl_xor(psum[r], mask, 32);
#pragma unroll
    for (int r = 0; r < 8; ++r) rowsum[r] = rowsum[r] * alpha[r] + psum[r];
#pragma unroll
    for (int i = 0; i < 4; ++i)
#pragma unroll
      for (int r = 0; r < 8; ++r) o[i][r] *= alpha[r];

    // reload P as A-fragment (same-wave DS ops are ordered)
    const bf16* pp = lb + ln * 40 + kh * 8;
    const v16bf pa = load_frag_bf16(pp, pp + 16);

    // ---- context += P[16x32] @ V[32x64] (V^T from LDS) ----
#pragma unroll
    for (int i = 0; i < 4; ++i) {
      const bf16* vp = &vbuf[cur][(i * 16 + ln) * VST + kh * 8];
      const v16bf vb = load_frag_bf16(vp, vp + 16);
      o[i] = wmma_bf16(pa, vb, o[i]);
    }

    wait_async_lds();   // next buffer fully landed in LDS
    __syncthreads();    // everyone done with current buffer
  }

  // ---- normalize, store context [b, t, h*64+d] ----
  float rinv[8];
#pragma unroll
  for (int r = 0; r < 8; ++r) rinv[r] = 1.0f / rowsum[r];
#pragma unroll
  for (int i = 0; i < 4; ++i)
#pragma unroll
    for (int r = 0; r < 8; ++r) {
      const int t = t0 + r + 8 * kh;
      ctx[((size_t)(b * T_SEQ + t) << 10) + h * D_K + i * 16 + ln] =
          (bf16)(o[i][r] * rinv[r]);
    }
}

// ---------------------------------------------------------------------------
// Kernel 3: output projection  out = ctx @ Wo^T + bo   (bf16 in, fp32 out)
// ---------------------------------------------------------------------------
__global__ __launch_bounds__(256) void out_proj_kernel(
    const bf16* __restrict__ Xc, const bf16* __restrict__ Wo,
    const float* __restrict__ bo, float* __restrict__ out) {
  const int lane   = threadIdx.x & 31;
  const int wave   = threadIdx.x >> 5;
  const int n_base = blockIdx.x * 128 + (wave & 3) * 32;
  const int m_base = blockIdx.y * 128 + (wave >> 2) * 64;
  const int ln = lane & 15, kh = lane >> 4;

  v8f acc[4][2] = {};
  for (int k0 = 0; k0 < D_MODEL; k0 += 32) {
    v16bf a[4], b[2];
#pragma unroll
    for (int i = 0; i < 4; ++i) {
      const bf16* pa = Xc + (size_t)(m_base + i * 16 + ln) * D_MODEL + k0 + kh * 8;
      a[i] = load_frag_bf16(pa, pa + 16);
    }
#pragma unroll
    for (int j = 0; j < 2; ++j) {
      const bf16* pb = Wo + (size_t)(n_base + j * 16 + ln) * D_MODEL + k0 + kh * 8;
      __builtin_prefetch(pb + 32, 0, 1);
      b[j] = load_frag_bf16(pb, pb + 16);
    }
#pragma unroll
    for (int i = 0; i < 4; ++i)
#pragma unroll
      for (int j = 0; j < 2; ++j) acc[i][j] = wmma_bf16(a[i], b[j], acc[i][j]);
  }

#pragma unroll
  for (int i = 0; i < 4; ++i)
#pragma unroll
    for (int j = 0; j < 2; ++j) {
      const int n_g = n_base + j * 16 + ln;
      const float bv = bo[n_g];
#pragma unroll
      for (int r = 0; r < 8; ++r) {
        const int m = m_base + i * 16 + r + 8 * kh;
        out[(size_t)m * D_MODEL + n_g] = acc[i][j][r] + bv;
      }
    }
}

// ---------------------------------------------------------------------------
extern "C" void kernel_launch(void* const* d_in, const int* in_sizes, int n_in,
                              void* d_out, int out_size, void* d_ws, size_t ws_size,
                              hipStream_t stream) {
  const float* query  = (const float*)d_in[0];
  const float* key_in = (const float*)d_in[1];
  const float* value  = (const float*)d_in[2];
  const float* Wq = (const float*)d_in[3];
  const float* bq = (const float*)d_in[4];
  const float* Wk = (const float*)d_in[5];
  const float* bk = (const float*)d_in[6];
  const float* Wv = (const float*)d_in[7];
  const float* bv = (const float*)d_in[8];
  const float* Wo = (const float*)d_in[9];
  const float* bo = (const float*)d_in[10];
  float* out = (float*)d_out;

  // Workspace: 7*8MiB activations + 4*2MiB weights = 64 MiB of bf16
  const size_t NIN = (size_t)2 * T_SEQ * D_MODEL;  // 4096*1024
  const size_t NW  = (size_t)D_MODEL * D_MODEL;
  bf16* p    = (bf16*)d_ws;
  bf16* Xq   = p; p += NIN;
  bf16* Xk   = p; p += NIN;
  bf16* Xv   = p; p += NIN;
  bf16* Wqb  = p; p += NW;
  bf16* Wkb  = p; p += NW;
  bf16* Wvb  = p; p += NW;
  bf16* Wob  = p; p += NW;
  bf16* Qws  = p; p += NIN;
  bf16* Kws  = p; p += NIN;
  bf16* Vtws = p; p += NIN;
  bf16* ctx  = p;

  // one-time fp32 -> bf16 conversions
  cvt_bf16_kernel<<<(int)(NIN / 8 / 256), 256, 0, stream>>>(query,  Xq, (int)(NIN / 8));
  cvt_bf16_kernel<<<(int)(NIN / 8 / 256), 256, 0, stream>>>(key_in, Xk, (int)(NIN / 8));
  cvt_bf16_kernel<<<(int)(NIN / 8 / 256), 256, 0, stream>>>(value,  Xv, (int)(NIN / 8));
  cvt_bf16_kernel<<<(int)(NW / 8 / 256),  256, 0, stream>>>(Wq, Wqb, (int)(NW / 8));
  cvt_bf16_kernel<<<(int)(NW / 8 / 256),  256, 0, stream>>>(Wk, Wkb, (int)(NW / 8));
  cvt_bf16_kernel<<<(int)(NW / 8 / 256),  256, 0, stream>>>(Wv, Wvb, (int)(NW / 8));
  cvt_bf16_kernel<<<(int)(NW / 8 / 256),  256, 0, stream>>>(Wo, Wob, (int)(NW / 8));

  const dim3 ggrid(D_MODEL / 128, (2 * T_SEQ) / 128);  // (8, 32)
  const float qscale = 0.125f;  // 1/sqrt(64)
  proj_kernel<<<ggrid, 256, 0, stream>>>(Xq, Wqb, bq, Qws, 0, qscale);
  proj_kernel<<<ggrid, 256, 0, stream>>>(Xk, Wkb, bk, Kws, 0, 1.0f);
  proj_kernel<<<ggrid, 256, 0, stream>>>(Xv, Wvb, bv, Vtws, 1, 1.0f);

  flash_attn_kernel<<<dim3(2 * H_NUM, T_SEQ / 128), 256, 0, stream>>>(Qws, Kws, Vtws, ctx);

  out_proj_kernel<<<ggrid, 256, 0, stream>>>(ctx, Wob, bo, out);
}
